// Decoder_7834020348031
// MI455X (gfx1250) — matrix operand order repaired and checked by reading, not compiled, not run on hardware
//
#include <hip/hip_runtime.h>
#include <hip/hip_bf16.h>

// Sizes from reference
#define Bsz 64
#define Tsz 64
#define Ssz 256
#define Hsz 1024
#define Esz 256
#define Vsz 32000
#define ZK  2304      // H + E + H  (z = [ctx, x_t, h])
#define G4  4096      // 4*H

typedef __attribute__((ext_vector_type(16))) __bf16 v16bf;
typedef __attribute__((ext_vector_type(8)))  __bf16 v8bf;
typedef __attribute__((ext_vector_type(8)))  float  v8f;

__device__ __forceinline__ float sigmoidf(float x) {
    return 1.0f / (1.0f + __expf(-x));
}

__device__ __forceinline__ v16bf load_a_frag(const __bf16* p) {
    v8bf lo = *(const v8bf*)(p);
    v8bf hi = *(const v8bf*)(p + 8);
    return __builtin_shufflevector(lo, hi,
            0, 1, 2, 3, 4, 5, 6, 7, 8, 9, 10, 11, 12, 13, 14, 15);
}

// ---------------------------------------------------------------------------
// One-time: fp32 -> bf16 transpose of [1024 x 32000] Wout into WoutT[32000][1024]
// ---------------------------------------------------------------------------
__global__ void transpose_f32_to_bf16(const float* __restrict__ src, __bf16* __restrict__ dst,
                                      int rows, int cols) {
    __shared__ float tile[32][33];
    int x  = blockIdx.x * 32 + threadIdx.x;         // src col
    int y0 = blockIdx.y * 32 + threadIdx.y;         // src row
#pragma unroll
    for (int j = 0; j < 32; j += 8) {
        int y = y0 + j;
        if (x < cols && y < rows)
            tile[threadIdx.y + j][threadIdx.x] = src[(size_t)y * cols + x];
    }
    __syncthreads();
    int x2 = blockIdx.y * 32 + threadIdx.x;         // dst col = src row
    int y2 = blockIdx.x * 32 + threadIdx.y;         // dst row = src col
#pragma unroll
    for (int j = 0; j < 32; j += 8) {
        int y = y2 + j;
        if (x2 < rows && y < cols)
            dst[(size_t)y * rows + x2] = (__bf16)tile[threadIdx.x][threadIdx.y + j];
    }
}

// ---------------------------------------------------------------------------
// One-time: build combined, transposed gate weight Wt[4096][2304] bf16 from
// Wx[1280][4096] (rows 0..1279) and Wh[1024][4096] (rows 1280..2303)
// ---------------------------------------------------------------------------
__global__ void transpose_gates_to_bf16(const float* __restrict__ Wx, const float* __restrict__ Wh,
                                        __bf16* __restrict__ dst) {
    __shared__ float tile[32][33];
    int x  = blockIdx.x * 32 + threadIdx.x;         // n (0..4095)
    int y0 = blockIdx.y * 32 + threadIdx.y;         // k (0..2303)
#pragma unroll
    for (int j = 0; j < 32; j += 8) {
        int k = y0 + j;
        float v;
        if (k < (Hsz + Esz)) v = Wx[(size_t)k * G4 + x];
        else                 v = Wh[(size_t)(k - (Hsz + Esz)) * G4 + x];
        tile[threadIdx.y + j][threadIdx.x] = v;
    }
    __syncthreads();
    int x2 = blockIdx.y * 32 + threadIdx.x;         // k
    int y2 = blockIdx.x * 32 + threadIdx.y;         // n
#pragma unroll
    for (int j = 0; j < 32; j += 8) {
        dst[(size_t)(y2 + j) * ZK + x2] = (__bf16)tile[threadIdx.x][threadIdx.y + j];
    }
}

// ---------------------------------------------------------------------------
// Per step: Luong attention (score -> softmax -> ctx) + pack z = [ctx, x_t, h]
// grid = B blocks, 256 threads (one thread per s for scoring)
// ---------------------------------------------------------------------------
__global__ void attn_build_z(const float* __restrict__ enc,      // [B,S,H]
                             const float* __restrict__ h,        // [B,H] f32
                             const int*   __restrict__ dec_in,   // [B,T]
                             const float* __restrict__ emb,      // [V,E]
                             __bf16* __restrict__ z,             // [B,ZK]
                             int t) {
    const int b   = blockIdx.x;
    const int tid = threadIdx.x;                    // 0..255
    __shared__ float hsh[Hsz];
    __shared__ float red[Ssz];
    __shared__ float satt[Ssz];

    const float* hrow = h + (size_t)b * Hsz;
    for (int i = tid; i < Hsz; i += 256) hsh[i] = hrow[i];
    __syncthreads();

    // score[s] = dot(enc[b,s,:], h[b,:])
    const float4* e4 = (const float4*)(enc + ((size_t)b * Ssz + tid) * Hsz);
    const float4* h4 = (const float4*)hsh;
    float sc = 0.0f;
#pragma unroll 4
    for (int k = 0; k < Hsz / 4; ++k) {
        float4 a = e4[k]; float4 c = h4[k];
        sc += a.x * c.x + a.y * c.y + a.z * c.z + a.w * c.w;
    }
    red[tid] = sc;
    __syncthreads();
    for (int off = 128; off > 0; off >>= 1) {
        if (tid < off) red[tid] = fmaxf(red[tid], red[tid + off]);
        __syncthreads();
    }
    const float mx = red[0];
    __syncthreads();
    const float ex = __expf(sc - mx);
    red[tid] = ex;
    __syncthreads();
    for (int off = 128; off > 0; off >>= 1) {
        if (tid < off) red[tid] += red[tid + off];
        __syncthreads();
    }
    const float inv_sum = 1.0f / red[0];
    satt[tid] = ex * inv_sum;
    __syncthreads();

    // ctx[hd] = sum_s att[s]*enc[b,s,hd]; thread owns 4 dims
    float4 acc = make_float4(0.f, 0.f, 0.f, 0.f);
    const float* ebase = enc + (size_t)b * Ssz * Hsz + tid * 4;
    for (int s = 0; s < Ssz; ++s) {
        const float a = satt[s];
        const float4 ev = *(const float4*)(ebase + (size_t)s * Hsz);
        acc.x += a * ev.x; acc.y += a * ev.y; acc.z += a * ev.z; acc.w += a * ev.w;
    }
    __bf16* zrow = z + (size_t)b * ZK;
    zrow[4 * tid + 0] = (__bf16)acc.x;
    zrow[4 * tid + 1] = (__bf16)acc.y;
    zrow[4 * tid + 2] = (__bf16)acc.z;
    zrow[4 * tid + 3] = (__bf16)acc.w;
    // x_t = emb[dec_in[b,t]]
    const int tok = dec_in[b * Tsz + t];
    zrow[Hsz + tid] = (__bf16)emb[(size_t)tok * Esz + tid];
    // h part
    zrow[Hsz + Esz + 4 * tid + 0] = (__bf16)hsh[4 * tid + 0];
    zrow[Hsz + Esz + 4 * tid + 1] = (__bf16)hsh[4 * tid + 1];
    zrow[Hsz + Esz + 4 * tid + 2] = (__bf16)hsh[4 * tid + 2];
    zrow[Hsz + Esz + 4 * tid + 3] = (__bf16)hsh[4 * tid + 3];
}

// ---------------------------------------------------------------------------
// WMMA bf16 GEMM: C[M,N](f32) = A[M,K](bf16, row-major) * Bt[N,K](bf16, K-major)
//                 + bias[N].
// One wave computes a 16x64 output slab (one M-tile x four N-tiles). The k-loop
// is software double-buffered: next iteration's fragments are loaded into fresh
// live values before this iteration's WMMAs issue, so the loads overlap the
// matrix ops (partial s_wait_loadcnt instead of full drains).
// N must be a multiple of 64, M of 16, K of 32.
// ---------------------------------------------------------------------------
__global__ void gemm_bf16_wmma(const __bf16* __restrict__ A,
                               const __bf16* __restrict__ Bt,
                               const float*  __restrict__ bias,
                               float* __restrict__ C,
                               int M, int N, int K, unsigned long long ldc) {
    const int wave    = threadIdx.x >> 5;
    const int slab_id = blockIdx.x * (blockDim.x >> 5) + wave;
    const int ngrp    = N >> 6;                       // 64-wide n slabs
    if (slab_id >= (M >> 4) * ngrp) return;           // whole wave exits; EXEC stays full
    const int m0 = (slab_id / ngrp) << 4;
    const int n0 = (slab_id % ngrp) << 6;

    const int lane = threadIdx.x & 31;
    const int half = lane >> 4;                       // 0: lanes 0-15, 1: lanes 16-31
    const int lr   = lane & 15;

    // A fragment: lane holds row m0+lr; 16 contiguous K starting at 0 (lo half) / 16 (hi half)
    const __bf16* Ap = A  + (size_t)(m0 + lr) * K + (half ? 16 : 0);
    // B fragment: lane holds col n+lr; 16 contiguous K starting at 0 (lo half) / 16 (hi half)
    const __bf16* Bp = Bt + (size_t)(n0 + lr) * K + (half ? 16 : 0);
    const size_t bs = (size_t)16 * K;                 // next 16-wide n tile

    v8f acc0 = {}, acc1 = {}, acc2 = {}, acc3 = {};

    // pipeline prologue: fragments for k = 0
    v16bf a_c  = load_a_frag(Ap);
    v16bf b_c0 = *(const v16bf*)(Bp);
    v16bf b_c1 = *(const v16bf*)(Bp + bs);
    v16bf b_c2 = *(const v16bf*)(Bp + 2 * bs);
    v16bf b_c3 = *(const v16bf*)(Bp + 3 * bs);

#pragma unroll 2
    for (int k = 32; k < K; k += 32) {
        // issue next iteration's loads first (distinct live values -> distinct
        // VGPRs -> WMMAs below only wait on the PREVIOUS iteration's loads)
        v16bf a_n  = load_a_frag(Ap + k);
        v16bf b_n0 = *(const v16bf*)(Bp + k);
        v16bf b_n1 = *(const v16bf*)(Bp + bs + k);
        v16bf b_n2 = *(const v16bf*)(Bp + 2 * bs + k);
        v16bf b_n3 = *(const v16bf*)(Bp + 3 * bs + k);

        acc0 = __builtin_amdgcn_wmma_f32_16x16x32_bf16(false, a_c, false, b_c0, (short)0, acc0, false, false);
        acc1 = __builtin_amdgcn_wmma_f32_16x16x32_bf16(false, a_c, false, b_c1, (short)0, acc1, false, false);
        acc2 = __builtin_amdgcn_wmma_f32_16x16x32_bf16(false, a_c, false, b_c2, (short)0, acc2, false, false);
        acc3 = __builtin_amdgcn_wmma_f32_16x16x32_bf16(false, a_c, false, b_c3, (short)0, acc3, false, false);

        a_c = a_n; b_c0 = b_n0; b_c1 = b_n1; b_c2 = b_n2; b_c3 = b_n3;
    }
    // pipeline epilogue: last k-step
    acc0 = __builtin_amdgcn_wmma_f32_16x16x32_bf16(false, a_c, false, b_c0, (short)0, acc0, false, false);
    acc1 = __builtin_amdgcn_wmma_f32_16x16x32_bf16(false, a_c, false, b_c1, (short)0, acc1, false, false);
    acc2 = __builtin_amdgcn_wmma_f32_16x16x32_bf16(false, a_c, false, b_c2, (short)0, acc2, false, false);
    acc3 = __builtin_amdgcn_wmma_f32_16x16x32_bf16(false, a_c, false, b_c3, (short)0, acc3, false, false);

    const int mbase = m0 + (half ? 8 : 0);
    v8f accs[4] = {acc0, acc1, acc2, acc3};
#pragma unroll
    for (int j = 0; j < 4; ++j) {
        const int n = n0 + 16 * j + lr;
        const float bb = bias ? bias[n] : 0.0f;
        float* Cp = C + (size_t)mbase * ldc + n;
#pragma unroll
        for (int v = 0; v < 8; ++v)
            Cp[(size_t)v * ldc] = accs[j][v] + bb;
    }
}

// ---------------------------------------------------------------------------
// LSTM cell pointwise: gates[B,4H] -> h,c (f32) and h (bf16)
// ---------------------------------------------------------------------------
__global__ void lstm_cell(const float* __restrict__ gates,
                          float* __restrict__ h, float* __restrict__ c,
                          __bf16* __restrict__ hbf) {
    const int idx = blockIdx.x * blockDim.x + threadIdx.x;   // B*H = 65536
    const int b = idx >> 10, j = idx & (Hsz - 1);
    const float* g = gates + (size_t)b * G4;
    const float iv = sigmoidf(g[j]);
    const float fv = sigmoidf(g[Hsz + j]);
    const float gv = tanhf(g[2 * Hsz + j]);
    const float ov = sigmoidf(g[3 * Hsz + j]);
    const float cn = fv * c[idx] + iv * gv;
    const float hn = ov * tanhf(cn);
    c[idx] = cn;
    h[idx] = hn;
    hbf[idx] = (__bf16)hn;
}

// ---------------------------------------------------------------------------
extern "C" void kernel_launch(void* const* d_in, const int* in_sizes, int n_in,
                              void* d_out, int out_size, void* d_ws, size_t ws_size,
                              hipStream_t stream) {
    const int*   dec_in = (const int*)  d_in[0];
    const float* enc    = (const float*)d_in[1];
    const float* dec_hs = (const float*)d_in[2];
    const float* dec_cs = (const float*)d_in[3];
    const float* emb    = (const float*)d_in[4];
    const float* Wx     = (const float*)d_in[5];
    const float* Wh     = (const float*)d_in[6];
    const float* bg     = (const float*)d_in[7];
    const float* Wout   = (const float*)d_in[8];
    const float* bout   = (const float*)d_in[9];
    float* out = (float*)d_out;

    // Workspace layout (all offsets 256B aligned)
    char* ws = (char*)d_ws;
    size_t off = 0;
    __bf16* WtG   = (__bf16*)(ws + off); off += (size_t)G4 * ZK * 2;            // 18.9 MB
    __bf16* WoutT = (__bf16*)(ws + off); off += (size_t)Vsz * Hsz * 2;          // 65.5 MB
    __bf16* z     = (__bf16*)(ws + off); off += (size_t)Bsz * ZK * 2;           // 295 KB
    float*  gates = (float*) (ws + off); off += (size_t)Bsz * G4 * 4;           // 1 MB
    float*  h_ws  = (float*) (ws + off); off += (size_t)Bsz * Hsz * 4;
    float*  c_ws  = (float*) (ws + off); off += (size_t)Bsz * Hsz * 4;
    __bf16* hbf   = (__bf16*)(ws + off); off += (size_t)Bsz * Hsz * 2;

    // init recurrent state
    hipMemcpyAsync(h_ws, dec_hs, (size_t)Bsz * Hsz * 4, hipMemcpyDeviceToDevice, stream);
    hipMemcpyAsync(c_ws, dec_cs, (size_t)Bsz * Hsz * 4, hipMemcpyDeviceToDevice, stream);

    // one-time bf16 weight conversion + transpose (keeps 64-step loop in L2)
    transpose_gates_to_bf16<<<dim3(G4 / 32, ZK / 32), dim3(32, 8), 0, stream>>>(Wx, Wh, WtG);
    transpose_f32_to_bf16<<<dim3(Vsz / 32, Hsz / 32), dim3(32, 8), 0, stream>>>(Wout, WoutT, Hsz, Vsz);

    const unsigned long long ldc_out = (unsigned long long)Tsz * Vsz;
    for (int t = 0; t < Tsz; ++t) {
        attn_build_z<<<Bsz, 256, 0, stream>>>(enc, h_ws, dec_in, emb, z, t);
        // gates = z @ [Wx;Wh] + b : M=64, N=4096, K=2304 -> 256 slab-waves / 32 blocks
        gemm_bf16_wmma<<<32, 256, 0, stream>>>(z, WtG, bg, gates,
                                               Bsz, G4, ZK, (unsigned long long)G4);
        lstm_cell<<<(Bsz * Hsz) / 256, 256, 0, stream>>>(gates, h_ws, c_ws, hbf);
        // logits = h_new @ Wout + bout : M=64, N=32000, K=1024 -> 2000 slab-waves / 250 blocks
        gemm_bf16_wmma<<<250, 256, 0, stream>>>(hbf, WoutT, bout, out + (size_t)t * Vsz,
                                                Bsz, Vsz, Hsz, ldc_out);
    }
}
